// FGMSA_15564961481362
// MI455X (gfx1250) — matrix operand order repaired and checked by reading, not compiled, not run on hardware
//
#include <hip/hip_runtime.h>
#include <hip/hip_bf16.h>
#include <math.h>

// ---------------------------------------------------------------------------
// Problem constants (from reference)
// ---------------------------------------------------------------------------
#define BB   8
#define HH   32
#define WW   32
#define NC   384
#define GROUPS 6
#define HEADS  12
#define HC   32            // per-head channels
#define GC   64            // per-group channels
#define NPOS (BB*HH*WW)    // 8192 total positions
#define NSAMP (HH*WW)      // 1024 tokens per batch
#define LN_EPS 1e-3f
#define ATT_SCALE 0.17677669529663689f   // 1/sqrt(32)

typedef __attribute__((ext_vector_type(16))) _Float16 v16h;
typedef __attribute__((ext_vector_type(8)))  _Float16 v8h;
typedef __attribute__((ext_vector_type(8)))  float    v8f;
typedef __attribute__((ext_vector_type(4)))  unsigned int u32x4;
typedef __attribute__((ext_vector_type(8)))  int          i32x8;
typedef __attribute__((ext_vector_type(4)))  int          i32x4;

// ---------------------------------------------------------------------------
// Tensor Data Mover: async 2D tile load Global(fp16) -> LDS.
// D# descriptor per CDNA5 ISA:
//  group0: [1:0]=count=1, [63:32]=lds_addr, [120:64]=global_addr, [127:126]=2
//  group1: data_size=1(2B), tensor_dim0/1, tile_dim0/1, tensor_dim0_stride
// Completion tracked with TENSORcnt (in-order per wave) -> software pipeline
// with s_wait_tensorcnt N.
// ---------------------------------------------------------------------------
__device__ __forceinline__ void tdm_load_2d_f16(unsigned lds_off,
                                                const void* gptr,
                                                unsigned tensor_d0,
                                                unsigned tensor_d1,
                                                unsigned tile_d0,
                                                unsigned tile_d1,
                                                unsigned stride_d0) {
    unsigned long long ga = (unsigned long long)gptr;
    u32x4 g0;
    g0[0] = 1u;                                        // count=1, user D#
    g0[1] = lds_off;                                   // LDS byte address
    g0[2] = (unsigned)(ga & 0xFFFFFFFFull);            // global addr lo
    g0[3] = (unsigned)((ga >> 32) & 0x1FFFFFFull) | (2u << 30); // hi | type=2
    i32x8 g1;
    g1[0] = (int)(1u << 16);                           // mask=0 | data_size=1 (2B)
    g1[1] = (int)(tensor_d0 << 16);                    // atomic_addr=0 | dim0 lo16
    g1[2] = (int)((tensor_d0 >> 16) | (tensor_d1 << 16));
    g1[3] = (int)((tensor_d1 >> 16) | (tile_d0 << 16));
    g1[4] = (int)(tile_d1 & 0xFFFFu);                  // tile_dim1 | tile_dim2=0
    g1[5] = (int)stride_d0;                            // dim0 stride lo32
    g1[6] = 0;
    g1[7] = 0;
    i32x4 z4 = {0, 0, 0, 0};
#if defined(__clang_major__) && __clang_major__ >= 23
    i32x8 z8 = {0, 0, 0, 0, 0, 0, 0, 0};
    __builtin_amdgcn_tensor_load_to_lds(g0, g1, z4, z4, z8, 0);
#else
    __builtin_amdgcn_tensor_load_to_lds(g0, g1, z4, z4, 0);
#endif
}

// 16-bit A-matrix (16x32) lane layout (CDNA5 ISA 7.12.2): per lane,
// elements 0..7 are K = 8*khalf..+7 (contiguous), elements 8..15 are
// K = 16+8*khalf..+7 (contiguous) -> two 16B vector loads + register concat.
__device__ __forceinline__ v16h load_a16(const _Float16* row, int khalf) {
    v8h lo = *(const v8h*)(row + khalf * 8);
    v8h hi = *(const v8h*)(row + 16 + khalf * 8);
    return __builtin_shufflevector(lo, hi, 0, 1, 2, 3, 4, 5, 6, 7,
                                   8, 9, 10, 11, 12, 13, 14, 15);
}

// ---------------------------------------------------------------------------
// fp32 -> fp16 conversion (one-time, bandwidth-trivial)
// ---------------------------------------------------------------------------
__global__ void cvt_f32_f16(const float* __restrict__ src,
                            _Float16* __restrict__ dst, int n) {
    int i = blockIdx.x * 256 + threadIdx.x;
    if (i < n) dst[i] = (_Float16)src[i];
}

// ---------------------------------------------------------------------------
// GEMM: Y[M,N] = Xh[M,K] @ Wh[K,N] + bias[N]. fp16 operands, fp32 accumulate.
// Block = 4 waves = 64x64 output tile; wave w owns rows w*16..w*16+15.
// Double-buffered TDM pipeline: wave 0 issues next (A,B) tile pair, then
// s_wait_tensorcnt(2) so only the in-flight pair remains outstanding.
// ---------------------------------------------------------------------------
template <bool F16OUT>
__global__ void gemm_bias_wmma(const _Float16* __restrict__ X,
                               const _Float16* __restrict__ Wt,
                               const float* __restrict__ bias,
                               void* __restrict__ Yv,
                               int M, int K, int N) {
    __shared__ __align__(16) _Float16 ldsA[2][64 * 32];  // [buf][row][k]
    __shared__ __align__(16) _Float16 ldsB[2][32 * 64];  // [buf][k][n]

    const int n0 = blockIdx.x * 64;
    const int m0 = blockIdx.y * 64;
    const int tid = threadIdx.x;
    const int wave = tid >> 5;
    const int lane = tid & 31;
    const int mrow = lane & 15;
    const int khalf = lane >> 4;
    const int nk = K >> 5;

    v8f acc[4] = {};

    if (wave == 0) {   // prologue: stage k-chunk 0 into buffer 0
        tdm_load_2d_f16((unsigned)(size_t)&ldsA[0][0], X + (size_t)m0 * K,
                        (unsigned)K, (unsigned)(M - m0), 32u, 64u, (unsigned)K);
        tdm_load_2d_f16((unsigned)(size_t)&ldsB[0][0], Wt + n0,
                        (unsigned)(N - n0), (unsigned)K, 64u, 32u, (unsigned)N);
    }

    for (int ki = 0; ki < nk; ++ki) {
        const int cur = ki & 1;
        if (wave == 0) {
            if (ki + 1 < nk) {        // issue next tile pair, keep 2 in flight
                const int k1 = (ki + 1) << 5;
                tdm_load_2d_f16((unsigned)(size_t)&ldsA[cur ^ 1][0],
                                X + (size_t)m0 * K + k1,
                                (unsigned)(K - k1), (unsigned)(M - m0),
                                32u, 64u, (unsigned)K);
                tdm_load_2d_f16((unsigned)(size_t)&ldsB[cur ^ 1][0],
                                Wt + (size_t)k1 * N + n0,
                                (unsigned)(N - n0), (unsigned)(K - k1),
                                64u, 32u, (unsigned)N);
                __builtin_amdgcn_s_wait_tensorcnt(2);  // current pair resident
            } else {
                __builtin_amdgcn_s_wait_tensorcnt(0);
            }
        }
        __syncthreads();

        v16h a = load_a16(&ldsA[cur][(wave * 16 + mrow) * 32], khalf);
        #pragma unroll
        for (int t = 0; t < 4; ++t) {
            v16h b;
            #pragma unroll
            for (int e = 0; e < 16; ++e)
                b[e] = ldsB[cur][(e < 8 ? khalf * 8 + e : 8 + khalf * 8 + e) * 64 + t * 16 + mrow];
            acc[t] = __builtin_amdgcn_wmma_f32_16x16x32_f16(
                         false, a, false, b, (short)0, acc[t], false, false);
        }
        __syncthreads();   // all waves done with buffer before TDM overwrites
    }

    // C/D layout: VGPR r -> row 8*khalf + r ; column = lane&15
    #pragma unroll
    for (int t = 0; t < 4; ++t) {
        #pragma unroll
        for (int r = 0; r < 8; ++r) {
            int m = m0 + wave * 16 + khalf * 8 + r;
            int n = n0 + t * 16 + mrow;
            float v = acc[t][r] + bias[n];
            if constexpr (F16OUT)
                ((_Float16*)Yv)[(size_t)m * N + n] = (_Float16)v;
            else
                ((float*)Yv)[(size_t)m * N + n] = v;
        }
    }
}

// ---------------------------------------------------------------------------
// Offset network: grouped 3x3 conv + LayerNorm + erf-GELU + per-group 64->2
// projection + tanh*16 + reference points. One block (384 thr) per pixel.
// ---------------------------------------------------------------------------
__global__ void offset_kernel(const _Float16* __restrict__ q,
                              const float* __restrict__ w_off0,  // [3,3,64,384]
                              const float* __restrict__ b_off0,
                              const float* __restrict__ ln_g,
                              const float* __restrict__ ln_b,
                              const float* __restrict__ w_offp,  // [64,2]
                              float* __restrict__ warp) {        // [B*G,1024,2]
    __shared__ float patch[9 * NC];
    __shared__ float red[NC];
    __shared__ float t_sh[NC];
    __shared__ float sh_mu, sh_rstd;

    const int p = blockIdx.x;                   // b*1024 + h*32 + w
    const int b = p >> 10, h = (p >> 5) & 31, w = p & 31;
    const int tid = threadIdx.x;

    for (int i = tid; i < 9 * NC; i += NC) {
        int dh = i / (3 * NC);
        int rem = i - dh * 3 * NC;
        int dw = rem / NC;
        int c  = rem - dw * NC;
        int hh = h + dh - 1, ww = w + dw - 1;
        float v = 0.f;
        if (hh >= 0 && hh < HH && ww >= 0 && ww < WW)
            v = (float)q[(size_t)((b * HH + hh) * WW + ww) * NC + c];
        patch[i] = v;
    }
    __syncthreads();

    const int oc = tid;
    const int g  = oc >> 6;
    float acc = b_off0[oc];
    #pragma unroll
    for (int tap = 0; tap < 9; ++tap) {
        const float* pin = patch + tap * NC + g * GC;
        const float* pw  = w_off0 + (size_t)(tap * GC) * NC + oc;
        #pragma unroll 8
        for (int ic = 0; ic < GC; ++ic)
            acc += pin[ic] * pw[(size_t)ic * NC];
    }

    red[tid] = acc;
    __syncthreads();
    if (tid == 0) {
        float s = 0.f;
        for (int i = 0; i < NC; ++i) s += red[i];
        sh_mu = s / (float)NC;
    }
    __syncthreads();
    float d = acc - sh_mu;
    red[tid] = d * d;
    __syncthreads();
    if (tid == 0) {
        float s = 0.f;
        for (int i = 0; i < NC; ++i) s += red[i];
        sh_rstd = rsqrtf(s / (float)NC + LN_EPS);
    }
    __syncthreads();

    float t = d * sh_rstd * ln_g[tid] + ln_b[tid];
    t = 0.5f * t * (1.f + erff(t * 0.70710678118654752f));   // exact erf GELU
    t_sh[tid] = t;
    __syncthreads();

    if (tid < GROUPS) {
        const float* ts = t_sh + tid * GC;
        float o0 = 0.f, o1 = 0.f;
        #pragma unroll 8
        for (int c = 0; c < GC; ++c) {
            o0 += ts[c] * w_offp[c * 2 + 0];
            o1 += ts[c] * w_offp[c * 2 + 1];
        }
        // reference meshgrid quirk: ref(h,w) = (w, h); warp = (pos1, pos0)
        float wx = tanhf(o1) * 16.f + (float)h;
        float wy = tanhf(o0) * 16.f + (float)w;
        size_t idx = ((size_t)(b * GROUPS + tid) * NSAMP + (h * WW + w)) * 2;
        warp[idx + 0] = wx;
        warp[idx + 1] = wy;
    }
}

// ---------------------------------------------------------------------------
// Bilinear sampling, zero border (pad-by-1 coordinate space). Writes fp16 xs.
// ---------------------------------------------------------------------------
__global__ void sample_kernel(const float* __restrict__ x,
                              const float* __restrict__ warp,
                              _Float16* __restrict__ xs) {
    const int qi = blockIdx.x;
    const int bg = blockIdx.y;
    const int b = bg / GROUPS, g = bg - b * GROUPS;
    const int c = threadIdx.x;

    size_t widx = ((size_t)bg * NSAMP + qi) * 2;
    float wx = warp[widx + 0] + 1.f;
    float wy = warp[widx + 1] + 1.f;
    float x0 = fminf(fmaxf(floorf(wx), 0.f), 32.f);
    float y0 = fminf(fmaxf(floorf(wy), 0.f), 32.f);
    float ax = fminf(fmaxf(wx - x0, 0.f), 1.f);
    float ay = fminf(fmaxf(wy - y0, 0.f), 1.f);
    int x0i = (int)x0, y0i = (int)y0;

    const float* xb = x + (size_t)b * HH * WW * NC + g * GC + c;
    auto fetch = [&](int yi, int xi) -> float {
        if (yi >= 1 && yi <= HH && xi >= 1 && xi <= WW)
            return xb[(size_t)((yi - 1) * WW + (xi - 1)) * NC];
        return 0.f;
    };
    float tl = fetch(y0i,     x0i);
    float tr = fetch(y0i,     x0i + 1);
    float bl = fetch(y0i + 1, x0i);
    float br = fetch(y0i + 1, x0i + 1);
    float top = tl + ax * (tr - tl);
    float bot = bl + ax * (br - bl);
    xs[((size_t)b * NSAMP + qi) * NC + g * GC + c] =
        (_Float16)(top + ay * (bot - top));
}

// ---------------------------------------------------------------------------
// Flash attention: block = 4 waves = 4 query tiles (64 queries) sharing one
// TDM-staged, double-buffered K chunk. Per 32-key chunk and wave:
// S = two WMMAs; online softmax; P via per-wave LDS -> A layout; O += P V.
// ---------------------------------------------------------------------------
__global__ void attn_kernel(const _Float16* __restrict__ qb,
                            const _Float16* __restrict__ kb,
                            const _Float16* __restrict__ vb,
                            _Float16* __restrict__ ob) {
    __shared__ __align__(16) _Float16 ldsK[2][32 * 32];  // [buf][key][ch]
    __shared__ __align__(16) _Float16 ldsP[4][16 * 32];  // [wave][row][key]

    const int h = blockIdx.y, b = blockIdx.z;
    const int tid  = threadIdx.x;
    const int wave = tid >> 5;
    const int lane = tid & 31;
    const int q0 = (blockIdx.x * 4 + wave) * 16;
    const int mrow  = lane & 15;
    const int khalf = lane >> 4;

    v16h aq = load_a16(qb + (size_t)(b * NSAMP + q0 + mrow) * NC + h * HC, khalf);

    v8f acc0 = {}, acc1 = {};
    float m_run[8], l_run[8];
    #pragma unroll
    for (int r = 0; r < 8; ++r) { m_run[r] = -3.0e38f; l_run[r] = 0.f; }

    if (wave == 0)   // prologue: stage key chunk 0
        tdm_load_2d_f16((unsigned)(size_t)&ldsK[0][0],
                        kb + (size_t)(b * NSAMP) * NC + h * HC,
                        32u, 32u, 32u, 32u, (unsigned)NC);

    for (int kt = 0; kt < NSAMP / 32; ++kt) {
        const int cur = kt & 1;
        if (wave == 0) {
            if (kt + 1 < NSAMP / 32) {     // pipeline next chunk
                tdm_load_2d_f16((unsigned)(size_t)&ldsK[cur ^ 1][0],
                                kb + (size_t)(b * NSAMP + (kt + 1) * 32) * NC + h * HC,
                                32u, 32u, 32u, 32u, (unsigned)NC);
                __builtin_amdgcn_s_wait_tensorcnt(1);
            } else {
                __builtin_amdgcn_s_wait_tensorcnt(0);
            }
        }
        __syncthreads();

        // B operands for S: lane%16 = key (N dim), element = channel (K, table order)
        v16h bk0 = load_a16(&ldsK[cur][mrow * 32], khalf);
        v16h bk1 = load_a16(&ldsK[cur][(16 + mrow) * 32], khalf);
        v8f z = {};
        v8f s0 = __builtin_amdgcn_wmma_f32_16x16x32_f16(
                     false, aq, false, bk0, (short)0, z, false, false);
        v8f s1 = __builtin_amdgcn_wmma_f32_16x16x32_f16(
                     false, aq, false, bk1, (short)0, z, false, false);

        // V chunk: lane%16 = channel, element = key (K, table order)
        const _Float16* vrow =
            vb + (size_t)(b * NSAMP + kt * 32) * NC + h * HC + mrow;
        v16h bv0, bv1;
        #pragma unroll
        for (int e = 0; e < 16; ++e) {
            const int kk = e < 8 ? khalf * 8 + e : 8 + khalf * 8 + e;
            bv0[e] = vrow[(size_t)kk * NC];
            bv1[e] = vrow[(size_t)kk * NC + 16];
        }

        // online softmax per row (row m = 8*khalf + r lives in one 16-lane half)
        #pragma unroll
        for (int r = 0; r < 8; ++r) {
            float t0 = s0[r] * ATT_SCALE;
            float t1 = s1[r] * ATT_SCALE;
            float mx = fmaxf(t0, t1);
            #pragma unroll
            for (int dlt = 1; dlt < 16; dlt <<= 1)
                mx = fmaxf(mx, __shfl_xor(mx, dlt, 32));
            float mnew = fmaxf(m_run[r], mx);
            float corr = __expf(m_run[r] - mnew);
            float p0 = __expf(t0 - mnew);
            float p1 = __expf(t1 - mnew);
            float rs = p0 + p1;
            #pragma unroll
            for (int dlt = 1; dlt < 16; dlt <<= 1)
                rs += __shfl_xor(rs, dlt, 32);
            l_run[r] = l_run[r] * corr + rs;
            m_run[r] = mnew;
            acc0[r] *= corr;
            acc1[r] *= corr;
            int m = khalf * 8 + r;
            ldsP[wave][m * 32 + mrow]      = (_Float16)p0;
            ldsP[wave][m * 32 + 16 + mrow] = (_Float16)p1;
        }
        __syncthreads();   // P visible; also fences K-buffer reads

        v16h pa = load_a16(&ldsP[wave][mrow * 32], khalf);
        acc0 = __builtin_amdgcn_wmma_f32_16x16x32_f16(
                   false, pa, false, bv0, (short)0, acc0, false, false);
        acc1 = __builtin_amdgcn_wmma_f32_16x16x32_f16(
                   false, pa, false, bv1, (short)0, acc1, false, false);
        __syncthreads();   // buffer swap safe
    }

    #pragma unroll
    for (int r = 0; r < 8; ++r) {
        int m = khalf * 8 + r;
        float inv = 1.f / l_run[r];
        _Float16* orow = ob + (size_t)(b * NSAMP + q0 + m) * NC + h * HC;
        orow[mrow]      = (_Float16)(acc0[r] * inv);
        orow[16 + mrow] = (_Float16)(acc1[r] * inv);
    }
}

// ---------------------------------------------------------------------------
// Launch
// ---------------------------------------------------------------------------
extern "C" void kernel_launch(void* const* d_in, const int* in_sizes, int n_in,
                              void* d_out, int out_size, void* d_ws, size_t ws_size,
                              hipStream_t stream) {
    (void)in_sizes; (void)n_in; (void)out_size; (void)ws_size;
    const float* x      = (const float*)d_in[0];
    const float* w_q    = (const float*)d_in[1];
    const float* b_q    = (const float*)d_in[2];
    const float* w_off0 = (const float*)d_in[3];
    const float* b_off0 = (const float*)d_in[4];
    const float* ln_g   = (const float*)d_in[5];
    const float* ln_b   = (const float*)d_in[6];
    const float* w_offp = (const float*)d_in[7];
    const float* w_k    = (const float*)d_in[8];
    const float* b_k    = (const float*)d_in[9];
    const float* w_v    = (const float*)d_in[10];
    const float* b_v    = (const float*)d_in[11];
    const float* w_o    = (const float*)d_in[12];
    const float* b_o    = (const float*)d_in[13];
    float* out = (float*)d_out;

    const size_t MAT = (size_t)NPOS * NC;      // 8192*384
    const size_t WMAT = (size_t)NC * NC;       // 384*384
    char* p = (char*)d_ws;
    auto carve = [&](size_t bytes) {
        char* r = p;
        p += (bytes + 255) & ~(size_t)255;
        return r;
    };
    _Float16* xh  = (_Float16*)carve(MAT * 2);
    _Float16* qh  = (_Float16*)carve(MAT * 2);
    _Float16* xsh = (_Float16*)carve(MAT * 2);
    _Float16* kh  = (_Float16*)carve(MAT * 2);
    _Float16* vh  = (_Float16*)carve(MAT * 2);
    _Float16* aoh = (_Float16*)carve(MAT * 2);
    _Float16* wqh = (_Float16*)carve(WMAT * 2);
    _Float16* wkh = (_Float16*)carve(WMAT * 2);
    _Float16* wvh = (_Float16*)carve(WMAT * 2);
    _Float16* woh = (_Float16*)carve(WMAT * 2);
    float* warp   = (float*)carve((size_t)BB * GROUPS * NSAMP * 2 * 4);

    // one-time fp32 -> fp16 conversions
    cvt_f32_f16<<<dim3((MAT + 255) / 256), dim3(256), 0, stream>>>(x, xh, (int)MAT);
    cvt_f32_f16<<<dim3((WMAT + 255) / 256), dim3(256), 0, stream>>>(w_q, wqh, (int)WMAT);
    cvt_f32_f16<<<dim3((WMAT + 255) / 256), dim3(256), 0, stream>>>(w_k, wkh, (int)WMAT);
    cvt_f32_f16<<<dim3((WMAT + 255) / 256), dim3(256), 0, stream>>>(w_v, wvh, (int)WMAT);
    cvt_f32_f16<<<dim3((WMAT + 255) / 256), dim3(256), 0, stream>>>(w_o, woh, (int)WMAT);

    dim3 gemm_grid(NC / 64, NPOS / 64);        // (6, 128)
    dim3 gemm_block(128);                      // 4 waves

    // q = x @ w_q + b_q   (fp16 out)
    gemm_bias_wmma<true><<<gemm_grid, gemm_block, 0, stream>>>(
        xh, wqh, b_q, qh, NPOS, NC, NC);
    // offset network -> warp grid
    offset_kernel<<<dim3(NPOS), dim3(NC), 0, stream>>>(
        qh, w_off0, b_off0, ln_g, ln_b, w_offp, warp);
    // bilinear sampling -> xs (fp16)
    sample_kernel<<<dim3(NSAMP, BB * GROUPS), dim3(GC), 0, stream>>>(x, warp, xsh);
    // k, v projections (fp16 out)
    gemm_bias_wmma<true><<<gemm_grid, gemm_block, 0, stream>>>(
        xsh, wkh, b_k, kh, NPOS, NC, NC);
    gemm_bias_wmma<true><<<gemm_grid, gemm_block, 0, stream>>>(
        xsh, wvh, b_v, vh, NPOS, NC, NC);
    // attention: 16 blocks of 4 query tiles per (head, batch)
    attn_kernel<<<dim3(NSAMP / 64, HEADS, BB), dim3(128), 0, stream>>>(
        qh, kh, vh, aoh);
    // output projection (fp32 out)
    gemm_bias_wmma<false><<<gemm_grid, gemm_block, 0, stream>>>(
        aoh, woh, b_o, out, NPOS, NC, NC);
}
